// LSTMModel_4209067950459
// MI455X (gfx1250) — matrix operand order repaired and checked by reading, not compiled, not run on hardware
//
#include <hip/hip_runtime.h>
#include <hip/hip_bf16.h>

// ---------------------------------------------------------------------------
// LSTM: B=256, T=512, I=256, H=1024.  Recurrent GEMM per step:
//   gates[256,4096] = [h | x_t][256,1280] @ W_cat^T[1280,4096]   (bf16 WMMA)
// A-operand staged to LDS with GLOBAL_LOAD_ASYNC_TO_LDS_B128 (double-buffered,
// overlapped with WMMA), then fused bias + activations + (c,h) update.
// ---------------------------------------------------------------------------

typedef __attribute__((ext_vector_type(16))) __bf16 v16bf;
typedef __attribute__((ext_vector_type(8)))  __bf16 v8bf;
typedef __attribute__((ext_vector_type(8)))  float  v8f;

#define BATCH   256
#define TSTEPS  512
#define ISZ     256
#define HSZ     1024
#define NGATES  4096            // 4*H
#define KTOT    1280            // H + I
#define NKT     40              // KTOT / 32  (K-tiles of 32)
#define NNT     256             // NGATES / 16 (N-tiles of 16)

// -------------------- weight pack: [nt][kt][lane][16 bf16] ------------------
// B-fragment (contiguous-K order, matching A reads): lane L holds column n = nt*16 + (L&15),
// K = kt*32 + (L>>4)*16 + e  for e = 0..15 (contiguous).
__global__ __launch_bounds__(128)
void pack_weights(const float* __restrict__ W_hh, const float* __restrict__ W_ih,
                  __bf16* __restrict__ Wpack) {
    int tid = blockIdx.x * 128 + threadIdx.x;      // one thread per (nt,kt,lane)
    if (tid >= NNT * NKT * 32) return;
    int nt   = tid / (NKT * 32);
    int rem  = tid % (NKT * 32);
    int kt   = rem / 32;
    int lane = rem & 31;
    int n     = nt * 16 + (lane & 15);
    int kbase = kt * 32 + ((lane >> 4) << 4);
    __bf16* dst = Wpack + (((size_t)(nt * NKT + kt)) * 32 + lane) * 16;
    #pragma unroll
    for (int e = 0; e < 16; ++e) {
        int k = kbase + e;
        float w = (k < HSZ) ? W_hh[(size_t)n * HSZ + k]
                            : W_ih[(size_t)n * ISZ + (k - HSZ)];
        dst[e] = (__bf16)w;
    }
}

__global__ void pack_bias(const float* __restrict__ b_ih, const float* __restrict__ b_hh,
                          float* __restrict__ bias) {
    int i = blockIdx.x * 128 + threadIdx.x;
    if (i < NGATES) bias[i] = b_ih[i] + b_hh[i];
}

__global__ __launch_bounds__(256)
void pack_x(const float* __restrict__ x, __bf16* __restrict__ xbf) {
    size_t i = ((size_t)blockIdx.x * 256 + threadIdx.x) * 8;
    #pragma unroll
    for (int e = 0; e < 8; ++e) xbf[i + e] = (__bf16)x[i + e];
}

__global__ void init_state(__bf16* h0, __bf16* h1, float* c) {
    int i = blockIdx.x * 256 + threadIdx.x;       // 256*1024 elements
    h0[i] = (__bf16)0.0f;
    h1[i] = (__bf16)0.0f;
    c[i]  = 0.0f;
}

// ------------------------------- LSTM step ----------------------------------
// grid = 256 WGs (8 batch-blocks x 32 hidden-col-blocks), 128 threads = 4 waves.
// wave g computes gate g's 32x32 tile for this (batch-block, hcol-block).
__global__ __launch_bounds__(128)
void lstm_step(const float* __restrict__ x, const __bf16* __restrict__ xbf,
               int use_xbf, const __bf16* __restrict__ Wpack,
               const float* __restrict__ bias, const __bf16* __restrict__ hin,
               __bf16* __restrict__ hout, float* __restrict__ c,
               float* __restrict__ hf, int t) {
    const int tid  = threadIdx.x;
    const int lane = tid & 31;
    const int wave = tid >> 5;                 // gate index 0..3 (i,f,g,o)
    const int m0   = (blockIdx.x & 7) * 32;    // batch rows
    const int n0   = (blockIdx.x >> 3) * 32;   // hidden cols

    __shared__ __align__(16) __bf16 As[2][32][320];  // double-buffered A (40 KB)
    __shared__ float Gl[4][32][32];                  // gate exchange     (16 KB)

    v8f acc[2][2] = {};                        // 2 M-tiles x 2 N-tiles, f32 C/D
    const int hk = lane >> 4;

    // stage phase `ph` (K span 320) of A = [h | x_t] into As[buf] (async path)
    auto stage = [&](int ph, int buf) {
        const int kph = ph * 320;
        #pragma unroll
        for (int q = 0; q < 10; ++q) {
            int cid  = tid + 128 * q;          // 1280 chunks of 8 bf16 (16 B)
            int row  = cid / 40;
            int kc   = (cid % 40) * 8;
            int k    = kph + kc;
            int brow = m0 + row;
            if (use_xbf || k < HSZ) {
                const __bf16* gp = (k < HSZ)
                    ? (hin + (size_t)brow * HSZ + k)
                    : (xbf + ((size_t)brow * TSTEPS + t) * ISZ + (k - HSZ));
                unsigned ldsoff = (unsigned)(size_t)(&As[buf][row][kc]);
                // async VGPR-bypass copy, tracked by ASYNCcnt
                asm volatile("global_load_async_to_lds_b128 %0, %1, off"
                             :: "v"(ldsoff), "v"(gp) : "memory");
            } else {                           // fallback: convert f32 x in-flight
                const float* xs = x + ((size_t)brow * TSTEPS + t) * ISZ + (k - HSZ);
                v8bf v;
                #pragma unroll
                for (int e = 0; e < 8; ++e) v[e] = (__bf16)xs[e];
                *(v8bf*)(&As[buf][row][kc]) = v;
            }
        }
    };

    stage(0, 0);
    asm volatile("s_wait_asynccnt 0x0" ::: "memory");
    __syncthreads();

    int cur = 0;
    #pragma unroll 1
    for (int ph = 0; ph < 4; ++ph) {           // K phases of 320
        if (ph + 1 < 4) stage(ph + 1, cur ^ 1);   // prefetch next phase (overlapped)

        #pragma unroll 2
        for (int it = 0; it < 10; ++it) {      // K-tiles of 32 within phase
            const int kloc = it * 32;
            const int ktG  = ph * 10 + it;
            // A fragments (contiguous K: lanes 0-15 K 0..15, lanes 16-31 K 16..31)
            v16bf a[2];
            #pragma unroll
            for (int mt = 0; mt < 2; ++mt) {
                int r = mt * 16 + (lane & 15);
                v8bf lo = *(const v8bf*)(&As[cur][r][kloc + hk * 16]);
                v8bf hi = *(const v8bf*)(&As[cur][r][kloc + hk * 16 + 8]);
                a[mt] = __builtin_shufflevector(lo, hi,
                        0,1,2,3,4,5,6,7,8,9,10,11,12,13,14,15);
            }
            #pragma unroll
            for (int ntl = 0; ntl < 2; ++ntl) {
                int ntG = wave * 64 + (n0 >> 4) + ntl;   // global N-tile of this gate
                const __bf16* bp = Wpack +
                    (((size_t)(ntG * NKT + ktG)) * 32 + lane) * 16;
                if (it + 1 < 10)                         // warm caches for next K-tile
                    __builtin_prefetch(bp + 32 * 16, 0, 3);
                v16bf b = *(const v16bf*)bp;
                #pragma unroll
                for (int mt = 0; mt < 2; ++mt)
                    acc[mt][ntl] = __builtin_amdgcn_wmma_f32_16x16x32_bf16(
                        false, a[mt], false, b, (short)0, acc[mt][ntl],
                        false, false);
            }
        }
        asm volatile("s_wait_asynccnt 0x0" ::: "memory");  // next-phase LDS writes done
        __syncthreads();
        cur ^= 1;
    }

    // spill gate tiles to LDS (C/D layout: VGPR v -> M = v + 8*(lane>=16), N = lane&15)
    #pragma unroll
    for (int mt = 0; mt < 2; ++mt)
        #pragma unroll
        for (int ntl = 0; ntl < 2; ++ntl) {
            int cc = ntl * 16 + (lane & 15);
            #pragma unroll
            for (int v = 0; v < 8; ++v) {
                int r = mt * 16 + v + (hk << 3);
                Gl[wave][r][cc] = acc[mt][ntl][v];
            }
        }
    __syncthreads();

    // fused bias + activations + state update: 1024 elems / 128 threads
    #pragma unroll
    for (int q = 0; q < 8; ++q) {
        int idx  = tid + 128 * q;
        int r    = idx >> 5;
        int cc   = idx & 31;
        int brow = m0 + r;
        int j    = n0 + cc;
        float ig = Gl[0][r][cc] + bias[j];
        float fg = Gl[1][r][cc] + bias[HSZ + j];
        float gg = Gl[2][r][cc] + bias[2 * HSZ + j];
        float og = Gl[3][r][cc] + bias[3 * HSZ + j];
        float is = 1.0f / (1.0f + __expf(-ig));
        float fs = 1.0f / (1.0f + __expf(-fg));
        float gt = tanhf(gg);
        float os = 1.0f / (1.0f + __expf(-og));
        size_t o = (size_t)brow * HSZ + j;
        float cn = fs * c[o] + is * gt;
        float hn = os * tanhf(cn);
        c[o]    = cn;
        hf[o]   = hn;                 // f32 copy for the output projection
        hout[o] = (__bf16)hn;         // bf16 for next step's GEMM
    }
}

// --------------------------- output projection ------------------------------
__global__ __launch_bounds__(256)
void out_proj(const float* __restrict__ hf, const float* __restrict__ Wout,
              const float* __restrict__ bout, float* __restrict__ out) {
    __shared__ float red[256];
    int b = blockIdx.x, tid = threadIdx.x;
    float s = 0.0f;
    for (int j = tid; j < HSZ; j += 256) s += hf[(size_t)b * HSZ + j] * Wout[j];
    red[tid] = s;
    __syncthreads();
    for (int off = 128; off > 0; off >>= 1) {
        if (tid < off) red[tid] += red[tid + off];
        __syncthreads();
    }
    if (tid == 0) out[b] = red[0] + bout[0];
}

// ----------------------------------------------------------------------------
extern "C" void kernel_launch(void* const* d_in, const int* in_sizes, int n_in,
                              void* d_out, int out_size, void* d_ws, size_t ws_size,
                              hipStream_t stream) {
    (void)in_sizes; (void)n_in; (void)out_size;
    const float* x     = (const float*)d_in[0];
    const float* W_ih  = (const float*)d_in[1];
    const float* W_hh  = (const float*)d_in[2];
    const float* b_ih  = (const float*)d_in[3];
    const float* b_hh  = (const float*)d_in[4];
    const float* W_out = (const float*)d_in[5];
    const float* b_out = (const float*)d_in[6];
    float* out = (float*)d_out;

    // workspace layout (all offsets multiple of 256B)
    char* ws = (char*)d_ws;
    __bf16* Wpack = (__bf16*)(ws);                               // 10,485,760 B
    float*  bias  = (float*) (ws + 10485760);                    //     16,384 B
    __bf16* h0    = (__bf16*)(ws + 10502144);                    //    524,288 B
    __bf16* h1    = (__bf16*)(ws + 11026432);                    //    524,288 B
    float*  cst   = (float*) (ws + 11550720);                    //  1,048,576 B
    float*  hf    = (float*) (ws + 12599296);                    //  1,048,576 B
    __bf16* xbf   = (__bf16*)(ws + 13647872);                    // 67,108,864 B
    const size_t need_xbf = 13647872ull + 67108864ull;
    int use_xbf = (ws_size >= need_xbf) ? 1 : 0;

    // 1) pack weights (bf16, WMMA B-fragment layout) + combined bias (+ bf16 x)
    pack_weights<<<(NNT * NKT * 32 + 127) / 128, 128, 0, stream>>>(W_hh, W_ih, Wpack);
    pack_bias<<<NGATES / 128, 128, 0, stream>>>(b_ih, b_hh, bias);
    if (use_xbf)
        pack_x<<<(BATCH * TSTEPS * ISZ) / (256 * 8), 256, 0, stream>>>(x, xbf);
    // 2) zero h (both ping-pong buffers) and c
    init_state<<<(BATCH * HSZ) / 256, 256, 0, stream>>>(h0, h1, cst);
    // 3) recurrence: 512 sequential fused GEMM+gate steps
    for (int t = 0; t < TSTEPS; ++t) {
        const __bf16* hin = (t & 1) ? h1 : h0;
        __bf16*      hout = (t & 1) ? h0 : h1;
        lstm_step<<<256, 128, 0, stream>>>(x, xbf, use_xbf, Wpack, bias,
                                           hin, hout, cst, hf, t);
    }
    // 4) out = h @ W_out^T + b_out
    out_proj<<<BATCH, 256, 0, stream>>>(hf, W_out, b_out, out);
}
